// MultiGCNLayers_61083024883811
// MI455X (gfx1250) — matrix operand order repaired and checked by reading, compiled, not run on hardware
//
#include <hip/hip_runtime.h>
#include <math.h>

#define DH 256  // hidden/output width (d_h == d_out == 256)

typedef __attribute__((ext_vector_type(16))) _Float16 v16h;
typedef __attribute__((ext_vector_type(8)))  _Float16 v8h;
typedef __attribute__((ext_vector_type(8)))  float    v8f;

// ---------------------------------------------------------------- utilities
__global__ __launch_bounds__(256) void k_fill0(float* __restrict__ p, long n) {
  long i = (long)blockIdx.x * blockDim.x + threadIdx.x;
  if (i < n) p[i] = 0.0f;
}

__global__ __launch_bounds__(256) void k_f2h(const float* __restrict__ in,
                                             _Float16* __restrict__ out, long n) {
  long i = (long)blockIdx.x * blockDim.x + threadIdx.x;
  if (i < n) out[i] = (_Float16)in[i];
}

// ------------------------------------------------------------ degree / norm
__global__ __launch_bounds__(256) void k_deg(const int* __restrict__ dst,
                                             float* __restrict__ deg, int E, int N) {
  int e = blockIdx.x * blockDim.x + threadIdx.x;
  if (e < E)          atomicAdd(&deg[dst[e]], 1.0f);
  else if (e < E + N) atomicAdd(&deg[e - E], 1.0f);   // self-loops
}

__global__ __launch_bounds__(256) void k_dinv(const float* __restrict__ deg,
                                              float* __restrict__ dinv, int N) {
  int i = blockIdx.x * blockDim.x + threadIdx.x;
  if (i < N) { float d = deg[i]; dinv[i] = (d > 0.0f) ? rsqrtf(d) : 0.0f; }
}

__global__ __launch_bounds__(256) void k_norm(const int* __restrict__ src,
                                              const int* __restrict__ dst,
                                              const float* __restrict__ dinv,
                                              float* __restrict__ norm, int E, int N) {
  int e = blockIdx.x * blockDim.x + threadIdx.x;
  if (e < E)          norm[e] = dinv[src[e]] * dinv[dst[e]];
  else if (e < E + N) { float v = dinv[e - E]; norm[e] = v * v; }
}

// ------------------------------------------------- weight repack for WMMA B
// Pack f32 W[K x 256] into f16 B-fragments: per 32x16 tile (kt, nt), 32 lanes
// x 16 halves contiguous. lane L, element j holds W[kt*32 + (j<8 ? (L>>4)*8 + j : 16 + (L>>4)*8 + j-8)][nt*16 + (L&15)].
__global__ __launch_bounds__(256) void k_packW(const float* __restrict__ W,
                                               _Float16* __restrict__ Bpk, int K) {
  int idx = blockIdx.x * blockDim.x + threadIdx.x;
  if (idx >= K * DH) return;
  int tile = idx >> 9, rem = idx & 511;
  int lane = rem >> 4, j = rem & 15;
  int kt = tile >> 4, nt = tile & 15;
  int kk = kt * 32 + (j < 8 ? (lane >> 4) * 8 + j : 16 + (lane >> 4) * 8 + (j - 8));
  int nn = nt * 16 + (lane & 15);
  Bpk[idx] = (_Float16)W[(size_t)kk * DH + nn];
}

// ------------------------------------------------------------ WMMA GEMM
// C[N x 256] = A[N x K](f16, row-major) @ W[K x 256] (pre-packed f16).
// grid.x = N/16 (N divisible by 16); block = 128 (4 waves); each wave owns a
// 16x64 output tile (4 col-tiles), reusing one A fragment per 4 WMMA issues.
__global__ __launch_bounds__(128) void k_gemm(const _Float16* __restrict__ A,
                                              const _Float16* __restrict__ Bpk,
                                              float* __restrict__ C, int K) {
  const int lane   = threadIdx.x & 31;
  const int wave   = threadIdx.x >> 5;
  const int row0   = blockIdx.x << 4;
  const int m      = lane & 15;
  const int khalf  = lane >> 4;
  const int ntBase = wave << 2;          // 4 col-tiles per wave

  v8f acc0 = {}, acc1 = {}, acc2 = {}, acc3 = {};
  const int ktiles = K >> 5;
  const _Float16* arowBase = A + (size_t)(row0 + m) * K + khalf * 8;
  const _Float16* bBase    = Bpk + (size_t)lane * 16;

  for (int kt = 0; kt < ktiles; ++kt) {
    // A fragment: a[0..7]=K[khalf*8..+7], a[8..15]=K[16+khalf*8..+7]
    v8h alo = *(const v8h*)(arowBase + kt * 32);
    v8h ahi = *(const v8h*)(arowBase + kt * 32 + 16);
    v16h a;
#pragma unroll
    for (int j = 0; j < 8; ++j) { a[j] = alo[j]; a[j + 8] = ahi[j]; }

    const _Float16* bt = bBase + ((size_t)(kt * 16 + ntBase) << 9);
    v16h b0 = *(const v16h*)(bt);
    v16h b1 = *(const v16h*)(bt + 512);
    v16h b2 = *(const v16h*)(bt + 1024);
    v16h b3 = *(const v16h*)(bt + 1536);

    acc0 = __builtin_amdgcn_wmma_f32_16x16x32_f16(false, a, false, b0, (short)0, acc0, false, false);
    acc1 = __builtin_amdgcn_wmma_f32_16x16x32_f16(false, a, false, b1, (short)0, acc1, false, false);
    acc2 = __builtin_amdgcn_wmma_f32_16x16x32_f16(false, a, false, b2, (short)0, acc2, false, false);
    acc3 = __builtin_amdgcn_wmma_f32_16x16x32_f16(false, a, false, b3, (short)0, acc3, false, false);
  }

  // C/D layout: lane -> col n = lane&15; rows mhalf*8 + r (r = VGPR index)
  const int n     = lane & 15;
  const int mhalf = lane >> 4;
  float* cBase = C + (size_t)(row0 + mhalf * 8) * DH + n;
  v8f accs[4] = {acc0, acc1, acc2, acc3};
#pragma unroll
  for (int t = 0; t < 4; ++t) {
    float* cp = cBase + (ntBase + t) * 16;
#pragma unroll
    for (int r = 0; r < 8; ++r) cp[(size_t)r * DH] = accs[t][r];
  }
}

// --------------------------------------------------- edge gather/scatter-add
// agg[dst] += lin[src] * norm ; 64 lanes x float4 per edge, 4 edges per block.
__global__ __launch_bounds__(256) void k_scatter(const float* __restrict__ lin,
                                                 float* __restrict__ agg,
                                                 const int* __restrict__ src,
                                                 const int* __restrict__ dst,
                                                 const float* __restrict__ norm,
                                                 int E, int N) {
  int e = blockIdx.x * 4 + (threadIdx.x >> 6);
  if (e >= E + N) return;
  int t = threadIdx.x & 63;
  int s, d;
  if (e < E) { s = src[e]; d = dst[e]; } else { s = e - E; d = s; }
  float w = norm[e];
  const float4 v = *(const float4*)(lin + (size_t)s * DH + t * 4);
  float* o = agg + (size_t)d * DH + t * 4;
  atomicAdd(o + 0, v.x * w);
  atomicAdd(o + 1, v.y * w);
  atomicAdd(o + 2, v.z * w);
  atomicAdd(o + 3, v.w * w);
}

// ------------------------------------------------ bias + ReLU residual
__global__ __launch_bounds__(256) void k_bias_act(const float* __restrict__ agg,
                                                  const float* __restrict__ b,
                                                  float* __restrict__ h, long total) {
  long i = (long)blockIdx.x * blockDim.x + threadIdx.x;
  if (i >= total) return;
  float t = agg[i] + b[i & (DH - 1)];
  h[i] = fmaxf(t, 0.0f) + t;          // relu(t) + t
}

// ------------------------------------------------------------- LayerNorm
__global__ __launch_bounds__(256) void k_layernorm(const float* __restrict__ h,
                                                   const float* __restrict__ lw,
                                                   const float* __restrict__ lb,
                                                   float* __restrict__ out, int N) {
  __shared__ float red[256];
  const int node = blockIdx.x, t = threadIdx.x;
  float v = h[(size_t)node * DH + t];
  red[t] = v; __syncthreads();
#pragma unroll
  for (int s = 128; s > 0; s >>= 1) { if (t < s) red[t] += red[t + s]; __syncthreads(); }
  float mean = red[0] * (1.0f / DH);
  __syncthreads();
  float dv = v - mean;
  red[t] = dv * dv; __syncthreads();
#pragma unroll
  for (int s = 128; s > 0; s >>= 1) { if (t < s) red[t] += red[t + s]; __syncthreads(); }
  float var = red[0] * (1.0f / DH);
  out[(size_t)node * DH + t] = dv * rsqrtf(var + 1e-6f) * lw[t] + lb[t];
}

__global__ __launch_bounds__(256) void k_batchs(const int* __restrict__ batch,
                                                float* __restrict__ out2, int N) {
  int i = blockIdx.x * blockDim.x + threadIdx.x;
  if (i < N) { float v = (float)batch[i]; out2[i] = v; out2[(size_t)N + i] = v; }
}

// ================================================================ launcher
extern "C" void kernel_launch(void* const* d_in, const int* in_sizes, int n_in,
                              void* d_out, int out_size, void* d_ws, size_t ws_size,
                              hipStream_t stream) {
  const float* x    = (const float*)d_in[0];
  const int*   edge = (const int*)d_in[1];
  const int*   batch= (const int*)d_in[2];
  const float* W0   = (const float*)d_in[3];
  const float* b0   = (const float*)d_in[4];
  const float* W1   = (const float*)d_in[5];
  const float* b1   = (const float*)d_in[6];
  const float* W2   = (const float*)d_in[7];
  const float* b2   = (const float*)d_in[8];
  const float* lnw  = (const float*)d_in[9];
  const float* lnb  = (const float*)d_in[10];

  const int N   = in_sizes[2];            // 50000 (divisible by 16)
  const int E   = in_sizes[1] / 2;        // 800000
  const int DIN = in_sizes[0] / N;        // 128
  const int EN  = E + N;
  const int* src = edge;
  const int* dst = edge + E;

  // workspace carve-out (256B aligned slices)
  char* ws = (char*)d_ws;
  size_t off = 0;
  auto take = [&](size_t bytes) { char* p = ws + off; off += (bytes + 255) & ~(size_t)255; return p; };
  float*    deg  = (float*)take((size_t)N * 4);
  float*    dinv = (float*)take((size_t)N * 4);
  float*    norm = (float*)take((size_t)EN * 4);
  _Float16* Ah   = (_Float16*)take((size_t)N * DH * 2);
  _Float16* Bpk  = (_Float16*)take((size_t)DH * DH * 2);
  float*    lin  = (float*)take((size_t)N * DH * 4);   // GEMM out, doubles as h
  float*    agg  = (float*)take((size_t)N * DH * 4);
  (void)ws_size; (void)n_in; (void)out_size;

  // symmetric normalization coefficients (shared by all layers/channels)
  k_fill0<<<(N + 255) / 256, 256, 0, stream>>>(deg, (long)N);
  k_deg  <<<(EN + 255) / 256, 256, 0, stream>>>(dst, deg, E, N);
  k_dinv <<<(N + 255) / 256, 256, 0, stream>>>(deg, dinv, N);
  k_norm <<<(EN + 255) / 256, 256, 0, stream>>>(src, dst, dinv, norm, E, N);

  float* out = (float*)d_out;
  const long nAgg = (long)N * DH;

  for (int c = 0; c < 2; ++c) {
    const float* Ws[3] = {W0 + (size_t)c * DIN * DH, W1 + (size_t)c * DH * DH, W2 + (size_t)c * DH * DH};
    const float* bs[3] = {b0 + c * DH, b1 + c * DH, b2 + c * DH};
    const int    Ks[3] = {DIN, DH, DH};
    for (int l = 0; l < 3; ++l) {
      const int K = Ks[l];
      const float* hin = (l == 0) ? x : lin;   // lin is consumed into Ah before GEMM overwrites it
      const long nA = (long)N * K;
      k_f2h  <<<(int)((nA + 255) / 256), 256, 0, stream>>>(hin, Ah, nA);
      k_packW<<<(K * DH + 255) / 256, 256, 0, stream>>>(Ws[l], Bpk, K);
      k_gemm <<<N / 16, 128, 0, stream>>>(Ah, Bpk, lin, K);
      k_fill0<<<(int)((nAgg + 255) / 256), 256, 0, stream>>>(agg, nAgg);
      k_scatter<<<(EN + 3) / 4, 256, 0, stream>>>(lin, agg, src, dst, norm, E, N);
      k_bias_act<<<(int)((nAgg + 255) / 256), 256, 0, stream>>>(agg, bs[l], lin, nAgg);
    }
    k_layernorm<<<N, 256, 0, stream>>>(lin, lnw, lnb, out + (size_t)c * N * DH, N);
  }
  k_batchs<<<(N + 255) / 256, 256, 0, stream>>>(batch, out + (size_t)2 * N * DH, N);
}
